// ReEigLayer_26018911879184
// MI455X (gfx1250) — compile-verified, matrix operand order checked by reading, not run on hardware
//
#include <hip/hip_runtime.h>
#include <hip/hip_bf16.h>

// ReEig for a batch of 64x64 symmetric matrices on gfx1250 (MI455X).
//  - one wave32 workgroup per matrix, whole problem resident in LDS
//  - parallel cyclic (round-robin) two-sided Jacobi, fixed 8 sweeps
//  - reconstruction V diag(max(lambda,1e-5)) V^T via f32 WMMA 16x16x4

#define LDA      68        // padded row stride (floats): rows 16B-aligned, banks spread
#define NSWEEPS  8
#define EIG_EPS  1e-05f

typedef __attribute__((ext_vector_type(2)))  float    v2f;
typedef __attribute__((ext_vector_type(8)))  float    v8f;
typedef __attribute__((ext_vector_type(16))) _Float16 v16h;

__global__ __launch_bounds__(32)
void reeig64_jacobi_wmma(const float* __restrict__ in, float* __restrict__ out, int batch)
{
    __shared__ float A[64 * LDA];   // working symmetric matrix
    __shared__ float V[64 * LDA];   // accumulated eigenvectors (columns)
    __shared__ float lam[64];       // clamped eigenvalues

    const int t = threadIdx.x;      // 0..31, wave32
    const int b = blockIdx.x;
    if (b >= batch) return;

    const float* __restrict__ Gin  = in  + (size_t)b * 4096;
    float*       __restrict__ Gout = out + (size_t)b * 4096;

    // ---- load + symmetrize (eigh uses (A + A^T)/2), V = I ----
    for (int idx = t; idx < 4096; idx += 32) {
        const int i = idx >> 6, j = idx & 63;
        A[i * LDA + j] = 0.5f * (Gin[i * 64 + j] + Gin[j * 64 + i]);
        V[i * LDA + j] = (i == j) ? 1.0f : 0.0f;
    }
    __syncthreads();

    // ---- parallel cyclic Jacobi: 63 rounds of 32 disjoint pairs per sweep ----
    for (int sweep = 0; sweep < NSWEEPS; ++sweep) {
        for (int r = 0; r < 63; ++r) {
            // round-robin tournament pairing: lane t owns pair (p,q)
            int p, q;
            if (t == 0) { p = 63; q = r; }
            else        { p = (r + t) % 63; q = (r + 63 - t) % 63; }

            const float app = A[p * LDA + p];
            const float aqq = A[q * LDA + q];
            const float apq = A[p * LDA + q];
            float c, s;
            if (__builtin_fabsf(apq) > 1e-30f) {
                const float tau = (aqq - app) / (2.0f * apq);
                const float tt  = __builtin_copysignf(1.0f, tau) /
                                  (__builtin_fabsf(tau) + __builtin_sqrtf(1.0f + tau * tau));
                c = 1.0f / __builtin_sqrtf(1.0f + tt * tt);
                s = tt * c;
            } else { c = 1.0f; s = 0.0f; }
            __syncthreads();

            // rows p,q: A <- G^T A   (rows are 16B-aligned: LDA*4 = 272 = 17*16)
            {
                float4* __restrict__ rp = reinterpret_cast<float4*>(&A[p * LDA]);
                float4* __restrict__ rq = reinterpret_cast<float4*>(&A[q * LDA]);
                #pragma unroll
                for (int j4 = 0; j4 < 16; ++j4) {
                    const float4 xp = rp[j4], xq = rq[j4];
                    float4 np, nq;
                    np.x = c * xp.x - s * xq.x;  nq.x = s * xp.x + c * xq.x;
                    np.y = c * xp.y - s * xq.y;  nq.y = s * xp.y + c * xq.y;
                    np.z = c * xp.z - s * xq.z;  nq.z = s * xp.z + c * xq.z;
                    np.w = c * xp.w - s * xq.w;  nq.w = s * xp.w + c * xq.w;
                    rp[j4] = np;  rq[j4] = nq;
                }
            }
            __syncthreads();

            // cols p,q of A (A <- A G) and of V (V <- V G)
            #pragma unroll 4
            for (int i = 0; i < 64; ++i) {
                const int ip = i * LDA + p, iq = i * LDA + q;
                const float xp = A[ip], xq = A[iq];
                A[ip] = c * xp - s * xq;
                A[iq] = s * xp + c * xq;
                const float vp = V[ip], vq = V[iq];
                V[ip] = c * vp - s * vq;
                V[iq] = s * vp + c * vq;
            }
            __syncthreads();
        }
    }

    // ---- clamp eigenvalues (diagonal of A) ----
    lam[t]      = fmaxf(A[t * LDA + t],               EIG_EPS);
    lam[t + 32] = fmaxf(A[(t + 32) * LDA + (t + 32)], EIG_EPS);
    __syncthreads();

    // ---- out = V diag(lam) V^T : 64x64x64 GEMM, 16x16 tiles on the matrix pipe ----
    const int m = t & 15;   // tile row (A-frag) / tile col (B-frag / D)
    const int h = t >> 4;   // half-wave selects K pair (f32) / K block (f16)

#if __has_builtin(__builtin_amdgcn_wmma_f32_16x16x4_f32)
    // exact-fp32 path: V_WMMA_F32_16X16X4_F32, K stepped by 4
    for (int ti = 0; ti < 4; ++ti) {
        for (int tj = 0; tj < 4; ++tj) {
            v8f acc = {};
            const float* __restrict__ Arow = &V[(ti * 16 + m) * LDA];
            const float* __restrict__ Brow = &V[(tj * 16 + m) * LDA];
            #pragma unroll 4
            for (int k4 = 0; k4 < 64; k4 += 4) {
                const int kb = k4 + 2 * h;                 // lanes 16-31 hold K+2 (ISA A layout)
                const float2 av = *reinterpret_cast<const float2*>(&Arow[kb]);
                const float2 bv = *reinterpret_cast<const float2*>(&Brow[kb]);
                v2f a, bb;
                a[0]  = av.x * lam[kb];                    // fold diag(lam) into A operand
                a[1]  = av.y * lam[kb + 1];
                bb[0] = bv.x;
                bb[1] = bv.y;
                acc = __builtin_amdgcn_wmma_f32_16x16x4_f32(
                          false, a, false, bb, (short)0, acc, false, false);
            }
            // D layout: VGPR rr -> row rr (lanes 0-15) / rr+8 (lanes 16-31), lane%16 -> col
            #pragma unroll
            for (int rr = 0; rr < 8; ++rr)
                Gout[(ti * 16 + rr + 8 * h) * 64 + tj * 16 + m] = acc[rr];
        }
    }
#else
    // fallback: codegen-confirmed V_WMMA_F32_16X16X32_F16, fragments per 16-bit A layout
    for (int ti = 0; ti < 4; ++ti) {
        for (int tj = 0; tj < 4; ++tj) {
            v8f acc = {};
            const float* __restrict__ Arow = &V[(ti * 16 + m) * LDA];
            const float* __restrict__ Brow = &V[(tj * 16 + m) * LDA];
            #pragma unroll
            for (int kb = 0; kb < 64; kb += 32) {
                v16h a, bb;
                #pragma unroll
                for (int v = 0; v < 8; ++v) {
                    // 16-bit A 16x32 layout: V0-3 -> K = 8h+2v{,+1}; V4-7 -> K = 16+8h+2(v-4){,+1}
                    const int k0 = kb + ((v < 4) ? (8 * h + 2 * v)
                                                 : (16 + 8 * h + 2 * (v - 4)));
                    a[2 * v]      = (_Float16)(Arow[k0]     * lam[k0]);
                    a[2 * v + 1]  = (_Float16)(Arow[k0 + 1] * lam[k0 + 1]);
                    bb[2 * v]     = (_Float16)Brow[k0];
                    bb[2 * v + 1] = (_Float16)Brow[k0 + 1];
                }
                acc = __builtin_amdgcn_wmma_f32_16x16x32_f16(
                          false, a, false, bb, (short)0, acc, false, false);
            }
            #pragma unroll
            for (int rr = 0; rr < 8; ++rr)
                Gout[(ti * 16 + rr + 8 * h) * 64 + tj * 16 + m] = acc[rr];
        }
    }
#endif
}

extern "C" void kernel_launch(void* const* d_in, const int* in_sizes, int n_in,
                              void* d_out, int out_size, void* d_ws, size_t ws_size,
                              hipStream_t stream) {
    (void)n_in; (void)out_size; (void)d_ws; (void)ws_size;
    const float* in  = (const float*)d_in[0];
    float*       out = (float*)d_out;
    const int batch = in_sizes[0] / 4096;   // 8192 matrices of 64*64
    reeig64_jacobi_wmma<<<dim3(batch), dim3(32), 0, stream>>>(in, out, batch);
}
